// AIAV_32212254720745
// MI455X (gfx1250) — matrix operand order repaired and checked by reading, not compiled, not run on hardware
//
#include <hip/hip_runtime.h>
#include <math.h>

#define SEQ   4096
#define INP   457
#define EMB   2048
#define GE    8192      // 4*EMB encoder gate rows
#define GD    1828      // 4*INP decoder gate rows
#define KXP   480       // padded input width (457 -> 480, mult of 32)
#define KE    2528      // KXP + EMB : fused [x;h] K for encoder
#define NWG_E 64
#define NWG_D 16

typedef __attribute__((ext_vector_type(8)))  __bf16 v8bf;
typedef __attribute__((ext_vector_type(16))) __bf16 v16bf;
typedef __attribute__((ext_vector_type(8)))  float  v8f;

__device__ __forceinline__ unsigned short f2bf(float f) {
  union { float f; unsigned u; } v; v.f = f;
  unsigned u = v.u;
  u += 0x7FFFu + ((u >> 16) & 1u);          // round-to-nearest-even
  return (unsigned short)(u >> 16);
}
__device__ __forceinline__ float bflo(unsigned u){ union{unsigned x;float f;}v; v.x = u << 16;        return v.f; }
__device__ __forceinline__ float bfhi(unsigned u){ union{unsigned x;float f;}v; v.x = u & 0xffff0000u; return v.f; }
__device__ __forceinline__ float bf2f(unsigned short h){ union{unsigned x;float f;}v; v.x = ((unsigned)h) << 16; return v.f; }
__device__ __forceinline__ float sigm(float x){ return 1.0f / (1.0f + __expf(-x)); }

// ---- atomic grid barrier (generation based, s_sleep backoff) ----
__device__ __forceinline__ void grid_sync(int* bar, int* cnt, int nwg) {
  __syncthreads();
  if (threadIdx.x == 0) {
    __threadfence();
    int gen = atomicAdd(bar, 0);
    if (atomicAdd(cnt, 1) == nwg - 1) {
      atomicExch(cnt, 0);
      atomicAdd(bar, 1);
    } else {
      while (atomicAdd(bar, 0) == gen) { __builtin_amdgcn_s_sleep(8); }
    }
  }
  __threadfence();
  __syncthreads();
}

// one 16x32 x 32x16 bf16 WMMA: weights tile (A, global) x vector-in-col0 (B, LDS)
#define WMMA_STEP(ACC, WPTR, BPTR)                                              \
  do {                                                                          \
    v8bf _a0 = *(const v8bf*)(WPTR);                                            \
    v8bf _a1 = *(const v8bf*)((WPTR) + 8);                                      \
    v16bf _av = __builtin_shufflevector(_a0, _a1,                               \
        0,1,2,3,4,5,6,7,8,9,10,11,12,13,14,15);                                 \
    v16bf _bv = *(const v16bf*)(BPTR);                                          \
    ACC = __builtin_amdgcn_wmma_f32_16x16x32_bf16(false, _av, false, _bv,       \
                                                  (short)0, ACC, false, false); \
  } while (0)

// =================== prep kernels (run once per launch) ===================
__global__ void init_state(unsigned short* hbuf, unsigned short* vdec, int* sync) {
  int i = blockIdx.x * blockDim.x + threadIdx.x;
  if (i < 2 * EMB) { hbuf[i] = 0; vdec[i] = 0; }
  if (i < 2) sync[i] = 0;
}

__global__ void prep_wcat(const float* __restrict__ Wih, const float* __restrict__ Whh,
                          unsigned short* __restrict__ Wcat, long n) {
  long i = (long)blockIdx.x * blockDim.x + threadIdx.x;
  if (i >= n) return;
  int r = (int)(i / KE);
  int k = (int)(i % KE);
  float v = 0.0f;
  if (k < INP)        v = Wih[(size_t)r * INP + k];
  else if (k >= KXP)  v = Whh[(size_t)r * EMB + (k - KXP)];
  Wcat[i] = f2bf(v);
}

__global__ void prep_xbf(const float* __restrict__ x, unsigned short* __restrict__ Xbf, long n) {
  long i = (long)blockIdx.x * blockDim.x + threadIdx.x;
  if (i >= n) return;
  int t = (int)(i / KXP);
  int k = (int)(i % KXP);
  Xbf[i] = f2bf(k < INP ? x[(size_t)t * INP + k] : 0.0f);
}

__global__ void prep_bias2(const float* __restrict__ b0, const float* __restrict__ b1,
                           float* __restrict__ out, int n) {
  int i = blockIdx.x * blockDim.x + threadIdx.x;
  if (i < n) out[i] = b0[i] + b1[i];
}

__global__ void prep_decw(const float* __restrict__ Wih, const float* __restrict__ Whh,
                          unsigned short* __restrict__ WihB, unsigned short* __restrict__ WsumB, long n) {
  long i = (long)blockIdx.x * blockDim.x + threadIdx.x;
  if (i >= n) return;
  float a = Wih[i], b = Whh[i];
  WihB[i]  = f2bf(a);
  WsumB[i] = f2bf(a + b);   // valid because xd == h for decoder steps t >= 1
}

__global__ void prep_cast(const float* __restrict__ src, unsigned short* __restrict__ dst, long n) {
  long i = (long)blockIdx.x * blockDim.x + threadIdx.x;
  if (i < n) dst[i] = f2bf(src[i]);
}

// =================== encoder: persistent, WMMA recurrence ===================
// 64 WGs x 256 threads. WG b owns h[b*32 .. +31]; 128 gate rows = 8 row-tiles
// of 16 (wave w -> tile w). gates = Wcat @ [x_t ; h] with the vector in tile
// column 0 (staged in LDS). All lanes load the same B data; accumulator
// columns 1..15 collect garbage -- legal since D columns are independent and
// only column 0 is read. Weights stream from L2 every step (41 MB resident).
__global__ __launch_bounds__(256) void enc_kernel(
    const unsigned short* __restrict__ Xbf,
    const unsigned short* __restrict__ Wcat,
    const float* __restrict__ bias,
    unsigned short* __restrict__ hbuf,
    int* bar, int* cnt)
{
  __shared__ __align__(32) unsigned short lds_v[KE];  // [x_t ; h] staged per step
  __shared__ float gl[128];     // gate values for this WG's 128 rows
  __shared__ float c_lds[32];   // private cell state chunk
  const int b    = blockIdx.x;
  const int tid  = threadIdx.x;
  const int lane = tid & 31;
  const int wv   = tid >> 5;          // 0..7 = row tile
  const int gate = wv >> 1;           // 0..3
  const int sub  = wv & 1;            // 0..1
  const int m    = gate * EMB + b * 32 + sub * 16 + (lane & 15);
  // A-matrix 16x32 bf16: lanes 0-15 hold K{0..15}, lanes 16-31 K{16..31}
  const unsigned short* Wrow = Wcat + (size_t)m * KE + ((lane >> 4) * 16);
  const unsigned short* Wh   = Wrow + KXP;
  const int boff = (lane >> 4) << 4;  // B: lanes 0-15 -> K+0, lanes 16-31 -> K+16
  const unsigned short* lvx = lds_v + boff;        // B source, x part
  const unsigned short* lvh = lds_v + KXP + boff;  // B source, h part
  if (tid < 32) c_lds[tid] = 0.0f;

  for (int t = 0; t < SEQ; ++t) {
    const int par = t & 1;
    // ---- stage [x_t ; h] into LDS: 60 + 256 uint4 ----
    {
      const uint4* xs4 = (const uint4*)(Xbf + (size_t)t * KXP);
      const uint4* hs4 = (const uint4*)(hbuf + par * EMB);
      for (int i = tid; i < (KXP + EMB) / 8; i += 256) {
        ((uint4*)lds_v)[i] = (i < KXP / 8) ? xs4[i] : hs4[i - KXP / 8];
      }
    }
    __syncthreads();
    v8f acc0 = {}, acc1 = {}, acc2 = {}, acc3 = {};
    // x part: 15 tiles = 3 x 4 + 3 straight-line (those 3 A-tiles get hoisted
    // into registers across the t loop -- cheap and beneficial)
    #pragma unroll 1
    for (int kt = 0; kt < 12; kt += 4) {
      WMMA_STEP(acc0, Wrow + (kt + 0) * 32, lvx + (kt + 0) * 32);
      WMMA_STEP(acc1, Wrow + (kt + 1) * 32, lvx + (kt + 1) * 32);
      WMMA_STEP(acc2, Wrow + (kt + 2) * 32, lvx + (kt + 2) * 32);
      WMMA_STEP(acc3, Wrow + (kt + 3) * 32, lvx + (kt + 3) * 32);
    }
    WMMA_STEP(acc0, Wrow + 12 * 32, lvx + 12 * 32);
    WMMA_STEP(acc1, Wrow + 13 * 32, lvx + 13 * 32);
    WMMA_STEP(acc2, Wrow + 14 * 32, lvx + 14 * 32);
    // h part: 64 tiles, 8 per iteration, 4 independent accumulator chains
    #pragma unroll 1
    for (int kt = 0; kt < EMB / 32; kt += 8) {
      WMMA_STEP(acc0, Wh + (kt + 0) * 32, lvh + (kt + 0) * 32);
      WMMA_STEP(acc1, Wh + (kt + 1) * 32, lvh + (kt + 1) * 32);
      WMMA_STEP(acc2, Wh + (kt + 2) * 32, lvh + (kt + 2) * 32);
      WMMA_STEP(acc3, Wh + (kt + 3) * 32, lvh + (kt + 3) * 32);
      WMMA_STEP(acc0, Wh + (kt + 4) * 32, lvh + (kt + 4) * 32);
      WMMA_STEP(acc1, Wh + (kt + 5) * 32, lvh + (kt + 5) * 32);
      WMMA_STEP(acc2, Wh + (kt + 6) * 32, lvh + (kt + 6) * 32);
      WMMA_STEP(acc3, Wh + (kt + 7) * 32, lvh + (kt + 7) * 32);
    }
    v8f acc = (acc0 + acc1) + (acc2 + acc3);
    // D column 0 lives in lane 0 (rows 0-7) and lane 16 (rows 8-15)
    if ((lane & 15) == 0) {
      const int base = wv * 16 + (lane >> 4) * 8;
      #pragma unroll
      for (int r = 0; r < 8; ++r) gl[base + r] = acc[r];
    }
    __syncthreads();
    if (tid < 32) {
      const int j = tid, s = j >> 4, idx = j & 15;
      const int col = b * 32 + j;
      float gi = gl[(0*2+s)*16 + idx] + bias[0*EMB + col];
      float gf = gl[(1*2+s)*16 + idx] + bias[1*EMB + col];
      float gg = gl[(2*2+s)*16 + idx] + bias[2*EMB + col];
      float go = gl[(3*2+s)*16 + idx] + bias[3*EMB + col];
      float c  = sigm(gf) * c_lds[j] + sigm(gi) * tanhf(gg);
      c_lds[j] = c;
      float h  = sigm(go) * tanhf(c);
      hbuf[(par ^ 1) * EMB + col] = f2bf(h);   // final h_enc ends in hbuf[0]
    }
    grid_sync(bar, cnt, NWG_E);
  }
}

// =================== decoder: persistent, latency-bound VALU ===================
__global__ __launch_bounds__(256) void dec_kernel(
    const unsigned short* __restrict__ henc,
    const unsigned short* __restrict__ Wih,
    const unsigned short* __restrict__ Wsum,
    const unsigned short* __restrict__ Whr,
    const float* __restrict__ bias_d,
    unsigned short* __restrict__ vdec,
    float* __restrict__ gates,
    float* __restrict__ cells,
    int* bar, int* cnt)
{
  __shared__ float c_l[INP];
  __shared__ float hr_l[INP];
  const int b    = blockIdx.x;
  const int tid  = threadIdx.x;
  const int gt   = b * 256 + tid;   // 0..4095
  const int row  = gt >> 1;
  const int half = gt & 1;
  for (int j = tid; j < INP; j += 256) c_l[j] = 0.0f;
  __syncthreads();

  for (int t = 0; t < SEQ; ++t) {
    const int par = t & 1;
    const unsigned short* v = (t == 0) ? henc : (vdec + par * EMB);
    const unsigned short* W = (t == 0) ? Wih  : Wsum;   // t==0: h=0 so only Wih
    float s = 0.0f;
    if (row < GD) {
      const unsigned short* wr = W + (size_t)row * EMB + half * 1024;
      const unsigned short* vv = v + half * 1024;
      #pragma unroll 4
      for (int k = 0; k < 1024; k += 8) {
        uint4 wq = *(const uint4*)(wr + k);
        uint4 vq = *(const uint4*)(vv + k);
        s += bflo(wq.x)*bflo(vq.x) + bfhi(wq.x)*bfhi(vq.x);
        s += bflo(wq.y)*bflo(vq.y) + bfhi(wq.y)*bfhi(vq.y);
        s += bflo(wq.z)*bflo(vq.z) + bfhi(wq.z)*bfhi(vq.z);
        s += bflo(wq.w)*bflo(vq.w) + bfhi(wq.w)*bfhi(vq.w);
      }
    }
    s += __shfl_xor(s, 1);
    if (row < GD && half == 0) gates[row] = s + bias_d[row];
    grid_sync(bar, cnt, NWG_D);
    for (int j = tid; j < INP; j += 256) {
      float gi = gates[j], gf = gates[INP + j], gg = gates[2*INP + j], go = gates[3*INP + j];
      float c  = sigm(gf) * c_l[j] + sigm(gi) * tanhf(gg);
      c_l[j] = c;
      if (b == 0) cells[(size_t)t * INP + j] = c;
      hr_l[j] = sigm(go) * tanhf(c);
    }
    __syncthreads();
    {
      const unsigned short* wr = Whr + (size_t)row * INP;
      const int k0 = half ? 228 : 0;
      const int k1 = half ? INP : 228;
      float sv = 0.0f;
      for (int k = k0; k < k1; ++k) sv += bf2f(wr[k]) * hr_l[k];
      sv += __shfl_xor(sv, 1);
      if (half == 0) vdec[(par ^ 1) * EMB + row] = f2bf(sv);
    }
    grid_sync(bar, cnt, NWG_D);
  }
}

// =================== softmax over reversed cells ===================
__global__ __launch_bounds__(256) void softmax_kernel(const float* __restrict__ cells,
                                                      float* __restrict__ out) {
  __shared__ float red[8];
  __shared__ float bc;
  const int r    = blockIdx.x;
  const float* src = cells + (size_t)(SEQ - 1 - r) * INP;
  float* dst       = out   + (size_t)r * INP;
  const int tid = threadIdx.x, lane = tid & 31, wv = tid >> 5;
  float m = -3.4e38f;
  for (int j = tid; j < INP; j += 256) m = fmaxf(m, src[j]);
  for (int o = 16; o > 0; o >>= 1) m = fmaxf(m, __shfl_xor(m, o));
  if (lane == 0) red[wv] = m;
  __syncthreads();
  if (tid == 0) { float v = red[0]; for (int i = 1; i < 8; ++i) v = fmaxf(v, red[i]); bc = v; }
  __syncthreads();
  m = bc;
  float s = 0.0f;
  for (int j = tid; j < INP; j += 256) s += __expf(src[j] - m);
  for (int o = 16; o > 0; o >>= 1) s += __shfl_xor(s, o);
  if (lane == 0) red[wv] = s;
  __syncthreads();
  if (tid == 0) { float v = 0.0f; for (int i = 0; i < 8; ++i) v += red[i]; bc = v; }
  __syncthreads();
  const float inv = 1.0f / bc;
  for (int j = tid; j < INP; j += 256) dst[j] = __expf(src[j] - m) * inv;
}

// =================== launcher ===================
extern "C" void kernel_launch(void* const* d_in, const int* in_sizes, int n_in,
                              void* d_out, int out_size, void* d_ws, size_t ws_size,
                              hipStream_t stream) {
  (void)in_sizes; (void)n_in; (void)out_size; (void)ws_size;
  const float* x       = (const float*)d_in[0];
  const float* enc_Wih = (const float*)d_in[1];
  const float* enc_Whh = (const float*)d_in[2];
  const float* enc_bih = (const float*)d_in[3];
  const float* enc_bhh = (const float*)d_in[4];
  const float* dec_Wih = (const float*)d_in[5];
  const float* dec_Whh = (const float*)d_in[6];
  const float* dec_bih = (const float*)d_in[7];
  const float* dec_bhh = (const float*)d_in[8];
  const float* dec_Whr = (const float*)d_in[9];

  char* ws = (char*)d_ws;
  size_t off = 0;
  auto alloc = [&](size_t bytes) -> char* {
    char* p = ws + off;
    off += (bytes + 255) & ~(size_t)255;
    return p;
  };
  int*            sync   = (int*)           alloc(256);
  unsigned short* Xbf    = (unsigned short*)alloc((size_t)SEQ * KXP * 2);
  unsigned short* Wcat   = (unsigned short*)alloc((size_t)GE  * KE  * 2);
  float*          bias_e = (float*)         alloc((size_t)GE * 4);
  unsigned short* hbuf   = (unsigned short*)alloc((size_t)2 * EMB * 2);
  unsigned short* WihD   = (unsigned short*)alloc((size_t)GD * EMB * 2);
  unsigned short* WsumD  = (unsigned short*)alloc((size_t)GD * EMB * 2);
  unsigned short* WhrB   = (unsigned short*)alloc((size_t)EMB * INP * 2);
  float*          bias_d = (float*)         alloc((size_t)GD * 4);
  unsigned short* vdec   = (unsigned short*)alloc((size_t)2 * EMB * 2);
  float*          gates  = (float*)         alloc((size_t)GD * 4);
  float*          cells  = (float*)         alloc((size_t)SEQ * INP * 4);

  init_state<<<(2 * EMB + 255) / 256, 256, 0, stream>>>(hbuf, vdec, sync);
  { long n = (long)GE * KE;
    prep_wcat<<<(int)((n + 255) / 256), 256, 0, stream>>>(enc_Wih, enc_Whh, Wcat, n); }
  { long n = (long)SEQ * KXP;
    prep_xbf<<<(int)((n + 255) / 256), 256, 0, stream>>>(x, Xbf, n); }
  prep_bias2<<<(GE + 255) / 256, 256, 0, stream>>>(enc_bih, enc_bhh, bias_e, GE);
  { long n = (long)GD * EMB;
    prep_decw<<<(int)((n + 255) / 256), 256, 0, stream>>>(dec_Wih, dec_Whh, WihD, WsumD, n); }
  { long n = (long)EMB * INP;
    prep_cast<<<(int)((n + 255) / 256), 256, 0, stream>>>(dec_Whr, WhrB, n); }
  prep_bias2<<<(GD + 255) / 256, 256, 0, stream>>>(dec_bih, dec_bhh, bias_d, GD);

  enc_kernel<<<NWG_E, 256, 0, stream>>>(Xbf, Wcat, bias_e, hbuf, sync, sync + 1);
  dec_kernel<<<NWG_D, 256, 0, stream>>>(hbuf /* h_enc = hbuf[0] */, WihD, WsumD, WhrB,
                                        bias_d, vdec, gates, cells, sync, sync + 1);
  softmax_kernel<<<SEQ, 256, 0, stream>>>(cells, (float*)d_out);
}
